// BDL_49606872269225
// MI455X (gfx1250) — matrix operand order repaired and checked by reading, not compiled, not run on hardware
//
#include <hip/hip_runtime.h>
#include <hip/hip_bf16.h>

// CDNA5 / gfx1250 fused "score all items" kernel.
//
// out[b,i] = sum_c softmax_c( sum_d u[b,d]*item[i,d]*W[c,d] + bias[c] ) * values[c]
//
// Per-class GEMM via v_wmma_f32_16x16x32_f16 with 5 lane-local accumulators.
// A fragments hold (u * W * log2e) in f16, built once per wave and reused
// across ~25 item tiles. Accumulators init to inline-0 (SRC2=0 in the WMMA);
// bias is folded multiplicatively into the softmax (exp(l+b)=exp(l)*exp(b)),
// and logits are pre-scaled by log2(e) so exp() is a single v_exp_f32 (2^x).
// Memory-bound: ~128 MB HBM traffic -> ~5.5 us floor @ 23.3 TB/s.

typedef __attribute__((ext_vector_type(16))) _Float16 v16h;
typedef __attribute__((ext_vector_type(8)))  float    v8f;

#define DIM    64
#define NITEM  100000
#define NCLS   5
#define ITILES (NITEM / 16)      // 6250, exact
#define WPB    8                 // waves per block (256 threads / wave32)
#define LOG2E  1.4426950408889634f

__global__ __launch_bounds__(256)
void fused_score_kernel(const int*   __restrict__ batch_user,
                        const float* __restrict__ user_table,
                        const float* __restrict__ item_table,
                        const float* __restrict__ cls_w,
                        const float* __restrict__ cls_b,
                        const float* __restrict__ values,
                        float*       __restrict__ out)
{
    const int lane     = threadIdx.x & 31;
    const int wave     = threadIdx.x >> 5;
    const int laneLo   = lane & 15;       // N (or M-row) within tile
    const int laneHalf = lane >> 4;       // 0: low half, 1: high half of wave
    const int userTile = blockIdx.y;      // 16 user tiles (256 users)

    // ---------------- Build A fragments: A[c] = (u * W[c] * log2e) in f16 ----
    // A 16x32 f16 layout (wave32): lanes 0-15 row M=lane hold K=0..7 (elts 0-7)
    // and K=16..23 (elts 8-15); lanes 16-31 row M=lane-16 hold K=8..15 / 24..31.
    const int    b    = userTile * 16 + laneLo;
    const int    urow = batch_user[b];
    const float* up   = user_table + (size_t)urow * DIM;

    v16h A[NCLS][2];
    #pragma unroll
    for (int kc = 0; kc < 2; ++kc) {
        const int d0 = kc * 32 + laneHalf * 8;
        float u0[8], u1[8];
        #pragma unroll
        for (int j = 0; j < 8; ++j) {
            u0[j] = up[d0 + j]      * LOG2E;   // pre-scale: logits in log2 domain
            u1[j] = up[d0 + 16 + j] * LOG2E;
        }
        #pragma unroll
        for (int c = 0; c < NCLS; ++c) {
            const float* wp = cls_w + c * DIM;
            #pragma unroll
            for (int j = 0; j < 8; ++j) {
                A[c][kc][j]     = (_Float16)(u0[j] * wp[d0 + j]);
                A[c][kc][8 + j] = (_Float16)(u1[j] * wp[d0 + 16 + j]);
            }
        }
    }

    // Bias folded multiplicatively: eb[c] = exp(bias[c]); ebv[c] = eb[c]*value[c]
    float eb[NCLS], ebv[NCLS];
    #pragma unroll
    for (int c = 0; c < NCLS; ++c) {
        eb[c]  = __expf(cls_b[c]);
        ebv[c] = eb[c] * values[c];
    }

    // C/D 16x16 f32 layout: VGPR e, lanes 0-15 -> M=e, lanes 16-31 -> M=e+8; N=laneLo.
    const int rowBase = userTile * 16 + laneHalf * 8;

    const v8f kZero = {};   // folds to inline SRC2=0 in the first WMMA

    // ---------------- Item-tile loop (grid-stride over 6250 tiles) -----------
    for (int it = blockIdx.x * WPB + wave; it < ITILES; it += gridDim.x * WPB) {
        const int itemBase = it * 16;

        // B 32x16 f16 layout: lane n (0-15) col N=n holds K=0..7 (elts 0-7) and
        // K=16..23 (elts 8-15); lane n+16 holds K=8..15 / 24..31 (matches A).
        const float* ip0 = item_table + (size_t)(itemBase + laneLo) * DIM
                         + laneHalf * 8;
        v16h B0, B1;
        #pragma unroll
        for (int j = 0; j < 16; ++j) B0[j] = (_Float16)ip0[j < 8 ? j : 8 + j];
        #pragma unroll
        for (int j = 0; j < 16; ++j) B1[j] = (_Float16)ip0[32 + (j < 8 ? j : 8 + j)];

        v8f acc[NCLS];
        #pragma unroll
        for (int c = 0; c < NCLS; ++c)
            acc[c] = __builtin_amdgcn_wmma_f32_16x16x32_f16(
                         false, A[c][0], false, B0, (short)0, kZero, false, false);
        #pragma unroll
        for (int c = 0; c < NCLS; ++c)
            acc[c] = __builtin_amdgcn_wmma_f32_16x16x32_f16(
                         false, A[c][1], false, B1, (short)0, acc[c], false, false);

        // ---- lane-local 5-class softmax + expected value ----
        // acc is log2(e)*logit, so 2^acc = exp(logit); |logits| are O(0.1),
        // so skipping max-subtraction is numerically safe.
        #pragma unroll
        for (int e = 0; e < 8; ++e) {
            float p0 = __builtin_amdgcn_exp2f(acc[0][e]);
            float p1 = __builtin_amdgcn_exp2f(acc[1][e]);
            float p2 = __builtin_amdgcn_exp2f(acc[2][e]);
            float p3 = __builtin_amdgcn_exp2f(acc[3][e]);
            float p4 = __builtin_amdgcn_exp2f(acc[4][e]);
            float s  = p0 * eb[0]  + p1 * eb[1]  + p2 * eb[2]
                     + p3 * eb[3]  + p4 * eb[4];
            float r  = p0 * ebv[0] + p1 * ebv[1] + p2 * ebv[2]
                     + p3 * ebv[3] + p4 * ebv[4];
            out[(size_t)(rowBase + e) * NITEM + itemBase + laneLo]
                = r * __builtin_amdgcn_rcpf(s);
        }
    }
}

extern "C" void kernel_launch(void* const* d_in, const int* in_sizes, int n_in,
                              void* d_out, int out_size, void* d_ws, size_t ws_size,
                              hipStream_t stream)
{
    const int*   batch_user = (const int*)  d_in[0];
    const float* user_table = (const float*)d_in[1];
    const float* item_table = (const float*)d_in[2];
    const float* cls_w      = (const float*)d_in[3];
    const float* cls_b      = (const float*)d_in[4];
    const float* values     = (const float*)d_in[5];
    float*       out        = (float*)d_out;

    // grid.x * WPB = 256 waves per user tile -> each wave does ~25 item tiles,
    // amortizing A-fragment construction; 4096 waves total fills the WGPs.
    dim3 grid(32, 16, 1);
    fused_score_kernel<<<grid, dim3(256, 1, 1), 0, stream>>>(
        batch_user, user_table, item_table, cls_w, cls_b, values, out);
}
